// SwinTransformerBlock_30597347016890
// MI455X (gfx1250) — compile-verified, matrix operand order checked by reading, not run on hardware
//
#include <hip/hip_runtime.h>
#include <hip/hip_bf16.h>
#include <math.h>

typedef __bf16 bf16_t;
typedef __attribute__((ext_vector_type(16))) __bf16 v16bf;
typedef __attribute__((ext_vector_type(8)))  float   v8f;

// ---------- helpers ----------
static __device__ __forceinline__ bf16_t to_bf16(float f) {
  unsigned u = __float_as_uint(f);
  u += 0x7FFFu + ((u >> 16) & 1u);                 // round-to-nearest-even
  unsigned short h = (unsigned short)(u >> 16);
  union { unsigned short u; bf16_t b; } c; c.u = h; return c.b;
}

// CDNA5 async global->LDS copy, 16B per lane. dst: low 32 bits of the flat
// shared-memory address (== LDS byte offset per the aperture truncation rule),
// src: 64-bit global address. Tracked on ASYNCcnt.
static __device__ __forceinline__ void async_copy_b128(void* lds_dst, const void* gsrc) {
  unsigned ldsaddr = (unsigned)(uintptr_t)lds_dst;
  unsigned long long gaddr = (unsigned long long)(uintptr_t)gsrc;
  asm volatile("global_load_async_to_lds_b128 %0, %1, off"
               :: "v"(ldsaddr), "v"(gaddr) : "memory");
}
static __device__ __forceinline__ void wait_asynccnt0() {
  asm volatile("s_wait_asynccnt 0x0" ::: "memory");
}

// Window-token index t (b, win, n) -> flat image index (b, h, w); identical
// mapping for forward gather (LN1+shift+partition) and reverse scatter.
static __device__ __forceinline__ int win_to_img(int t) {
  int b = t / 3136; int r = t - b * 3136;
  int win = r / 49;  int n = r - win * 49;
  int wh = win >> 3, ww = win & 7;
  int i = n / 7, j = n - i * 7;
  int h = wh * 7 + i + 3; if (h >= 56) h -= 56;
  int w = ww * 7 + j + 3; if (w >= 56) w -= 56;
  return (b * 56 + h) * 56 + w;
}

// Row-major 16-bit fragment (A-style; also B if the tile is stored transposed
// N x K). Per ISA 7.12.2: per-lane data = elements [0..7] and [16..23] of the
// lane's row (offset half*8) -> two 16B contiguous LDS loads.
static __device__ __forceinline__ v16bf load_frag_row(const bf16_t* p, int ld) {
  const int lane = threadIdx.x & 31;
  const bf16_t* rp = p + (lane & 15) * ld + (lane >> 4) * 8;
  union { uint4 q[2]; v16bf v; } u;
  u.q[0] = *(const uint4*)rp;          // K elements 0..7
  u.q[1] = *(const uint4*)(rp + 16);   // K elements 16..23
  return u.v;
}

#define WMMA_BF16(A, B, C) \
  __builtin_amdgcn_wmma_f32_16x16x32_bf16(false, (A), false, (B), (short)0, (C), false, false)

// ---------- fp32 -> bf16 weight conversion ----------
__global__ __launch_bounds__(256) void cvt_kernel(const float* __restrict__ s,
                                                  bf16_t* __restrict__ d, int n) {
  int i = blockIdx.x * 256 + threadIdx.x;
  if (i < n) d[i] = to_bf16(s[i]);
}

// ---------- fused rel-pos-bias + shifted-window-mask table ----------
// bm[cls][head][n1][n2], cls = (wh==7)*2 + (ww==7); padded n>=49 -> -1e30.
__global__ __launch_bounds__(256) void build_bm_kernel(const float* __restrict__ btab,
                                                       float* __restrict__ bm) {
  int idx = blockIdx.x * 256 + threadIdx.x;    // 4*16*64*64 = 262144
  int n2 = idx & 63;
  int n1 = (idx >> 6) & 63;
  int head = (idx >> 12) & 15;
  int cls = idx >> 16;
  float v;
  if (n1 < 49 && n2 < 49) {
    int i1 = n1 / 7, j1 = n1 % 7, i2 = n2 / 7, j2 = n2 % 7;
    int rpi = (i1 - i2 + 6) * 13 + (j1 - j2 + 6);
    v = btab[rpi * 16 + head];
    int he = cls >> 1, we = cls & 1;
    int rh1 = he ? (i1 < 4 ? 1 : 2) : 0;
    int rw1 = we ? (j1 < 4 ? 1 : 2) : 0;
    int rh2 = he ? (i2 < 4 ? 1 : 2) : 0;
    int rw2 = we ? (j2 < 4 ? 1 : 2) : 0;
    if ((rh1 * 3 + rw1) != (rh2 * 3 + rw2)) v -= 100.0f;
  } else {
    v = -1e30f;
  }
  bm[idx] = v;
}

// ---------- LayerNorm (optionally fused with shift+window-partition gather) ----------
__global__ __launch_bounds__(256) void ln_kernel(const float* __restrict__ x,
                                                 const float* __restrict__ gw,
                                                 const float* __restrict__ gb,
                                                 bf16_t* __restrict__ out, int gather) {
  __shared__ float red[256];
  const int t = blockIdx.x;
  const int tid = threadIdx.x;
  const int src = gather ? win_to_img(t) : t;
  const float* row = x + (size_t)src * 512;
  float v0 = row[tid], v1 = row[tid + 256];
  red[tid] = v0 + v1;
  __syncthreads();
  for (int off = 128; off > 0; off >>= 1) { if (tid < off) red[tid] += red[tid + off]; __syncthreads(); }
  float mean = red[0] * (1.0f / 512.0f);
  __syncthreads();
  float d0 = v0 - mean, d1 = v1 - mean;
  red[tid] = d0 * d0 + d1 * d1;
  __syncthreads();
  for (int off = 128; off > 0; off >>= 1) { if (tid < off) red[tid] += red[tid + off]; __syncthreads(); }
  float rstd = rsqrtf(red[0] * (1.0f / 512.0f) + 1e-5f);
  size_t o = (size_t)t * 512;
  out[o + tid]       = to_bf16(d0 * rstd * gw[tid] + gb[tid]);
  out[o + tid + 256] = to_bf16(d1 * rstd * gw[tid + 256] + gb[tid + 256]);
}

// ---------- tiled bf16 WMMA GEMM: C[MxN] = A[MxK] * B[KxN] + bias ----------
// 128x128 tile, 8 waves of 32x64, K-step 32, double-buffered LDS (one barrier
// per step). A staged via async global->LDS (ASYNCcnt); B staged transposed
// (N x K) through registers so all fragments are b128 LDS loads.
// EPI 0: store bf16.  EPI 1: GELU, store bf16.
// EPI 2: proj: scatter window->image, add residual (x), store fp32.
// EPI 3: fc2: add residual (x1), store fp32.
template <int EPI>
__global__ __launch_bounds__(256) void gemm_bf16_kernel(
    const bf16_t* __restrict__ A, const bf16_t* __restrict__ B,
    const float* __restrict__ bias, int M, int N, int K,
    bf16_t* __restrict__ outB, float* __restrict__ outF,
    const float* __restrict__ resid) {
  __shared__ __align__(16) bf16_t As[2][128 * 40];   // 128 rows x 32 K (+8 pad)
  __shared__ __align__(16) bf16_t Bt[2][128 * 40];   // 128 cols x 32 K (+8 pad), transposed
  const int tid = threadIdx.x;
  const int m0 = blockIdx.x * 128;
  const int n0 = blockIdx.y * 128;
  const int wave = tid >> 5;
  const int lane = tid & 31;
  const int wm = wave >> 1;        // 0..3 : 32-row strip
  const int wn = wave & 1;         // 0..1 : 64-col strip
  const int half = lane >> 4;
  const int lc = lane & 15;

  v8f acc[2][4] = {};

  const int ar = tid >> 1;               // 0..127 (A row)
  const int ac = (tid & 1) * 16;         // 0 / 16 (A col half)
  const int br = tid >> 3;               // 0..31  (B k-row)
  const int bc = (tid & 7) * 16;         // 0..112 (B col base)

  // prologue: async A tile 0, synchronous transposed B tile 0
  async_copy_b128(&As[0][ar * 40 + ac],     A + (size_t)(m0 + ar) * K + ac);
  async_copy_b128(&As[0][ar * 40 + ac + 8], A + (size_t)(m0 + ar) * K + ac + 8);
  uint4 b0, b1;
  {
    const uint4* gb = (const uint4*)(B + (size_t)br * N + n0 + bc);
    b0 = gb[0]; b1 = gb[1];
    union { uint4 u; bf16_t h[8]; } u0, u1; u0.u = b0; u1.u = b1;
#pragma unroll
    for (int u = 0; u < 8; ++u) {
      Bt[0][(bc + u) * 40 + br]     = u0.h[u];
      Bt[0][(bc + 8 + u) * 40 + br] = u1.h[u];
    }
  }
  wait_asynccnt0();
  __syncthreads();

  int p = 0;
  for (int k0 = 0; k0 < K; k0 += 32) {
    const bool nxt = (k0 + 32) < K;
    if (nxt) {
      // issue next A tile asynchronously; fetch next B slab into registers
      async_copy_b128(&As[p ^ 1][ar * 40 + ac],
                      A + (size_t)(m0 + ar) * K + k0 + 32 + ac);
      async_copy_b128(&As[p ^ 1][ar * 40 + ac + 8],
                      A + (size_t)(m0 + ar) * K + k0 + 32 + ac + 8);
      const uint4* gb = (const uint4*)(B + (size_t)(k0 + 32 + br) * N + n0 + bc);
      b0 = gb[0]; b1 = gb[1];
      if (k0 + 64 < K)  // CDNA5 global_prefetch of the K slab after next
        __builtin_prefetch((const void*)(A + (size_t)(m0 + ar) * K + k0 + 64 + ac), 0, 1);
    }
    v16bf fa[2], fb[4];
    fa[0] = load_frag_row(&As[p][(wm * 32) * 40], 40);
    fa[1] = load_frag_row(&As[p][(wm * 32 + 16) * 40], 40);
#pragma unroll
    for (int j = 0; j < 4; ++j)
      fb[j] = load_frag_row(&Bt[p][(wn * 64 + j * 16) * 40], 40);
#pragma unroll
    for (int i = 0; i < 2; ++i)
#pragma unroll
      for (int j = 0; j < 4; ++j)
        acc[i][j] = WMMA_BF16(fa[i], fb[j], acc[i][j]);
    if (nxt) {
      union { uint4 u; bf16_t h[8]; } u0, u1; u0.u = b0; u1.u = b1;
#pragma unroll
      for (int u = 0; u < 8; ++u) {
        Bt[p ^ 1][(bc + u) * 40 + br]     = u0.h[u];
        Bt[p ^ 1][(bc + 8 + u) * 40 + br] = u1.h[u];
      }
    }
    wait_asynccnt0();
    __syncthreads();
    p ^= 1;
  }

#pragma unroll
  for (int j = 0; j < 4; ++j) {
    const int col = n0 + wn * 64 + j * 16 + lc;
    const float bcol = bias[col];
#pragma unroll
    for (int i = 0; i < 2; ++i) {
#pragma unroll
      for (int v = 0; v < 8; ++v) {
        int row = m0 + wm * 32 + i * 16 + v + half * 8;
        float val = acc[i][j][v] + bcol;
        if (EPI == 0) {
          outB[(size_t)row * N + col] = to_bf16(val);
        } else if (EPI == 1) {
          val = 0.5f * val * (1.0f + erff(val * 0.70710678118654752f));
          outB[(size_t)row * N + col] = to_bf16(val);
        } else if (EPI == 2) {
          size_t idx = (size_t)win_to_img(row) * 512 + col;
          outF[idx] = resid[idx] + val;
        } else {
          size_t idx = (size_t)row * 512 + col;
          outF[idx] = resid[idx] + val;
        }
      }
    }
  }
}

// ---------- windowed attention: one wave per (window, head) ----------
__global__ __launch_bounds__(32) void attn_kernel(const bf16_t* __restrict__ qkv,
                                                  const float* __restrict__ bm,
                                                  bf16_t* __restrict__ out) {
  __shared__ __align__(16) bf16_t q_sh[64 * 32];
  __shared__ __align__(16) bf16_t k_sh[64 * 32];
  __shared__ __align__(16) bf16_t vT_sh[32 * 72];  // transposed: [d][k(+8 pad)]
  __shared__ float  s_sh[64 * 65];
  __shared__ __align__(16) bf16_t p_sh[64 * 64];

  const int wb = blockIdx.x >> 4;      // window-batch [0, 2048)
  const int head = blockIdx.x & 15;
  const int lane = threadIdx.x;
  const int half = lane >> 4;
  const int lc = lane & 15;
  const int win = wb & 63;
  const int cls = ((win >> 3) == 7 ? 2 : 0) | ((win & 7) == 7 ? 1 : 0);
  const float* bmp = bm + ((size_t)(cls * 16 + head)) * 4096;  // [64][64]

  // stage q/k rows asynchronously; v transposed synchronously (zero-pad to 64)
  for (int r = lane; r < 64; r += 32) {
    if (r < 49) {
      const size_t base = (size_t)(wb * 49 + r) * 1536 + head * 32;
#pragma unroll
      for (int u = 0; u < 4; ++u) {
        async_copy_b128(q_sh + r * 32 + u * 8, qkv + base + u * 8);
        async_copy_b128(k_sh + r * 32 + u * 8, qkv + base + 512 + u * 8);
      }
      union { uint4 u[4]; bf16_t h[32]; } vu;
      const uint4* gv = (const uint4*)(qkv + base + 1024);
#pragma unroll
      for (int u = 0; u < 4; ++u) vu.u[u] = gv[u];
#pragma unroll
      for (int d = 0; d < 32; ++d) vT_sh[d * 72 + r] = vu.h[d];
    } else {
      uint4 z = make_uint4(0, 0, 0, 0);
#pragma unroll
      for (int u = 0; u < 4; ++u) {
        ((uint4*)(q_sh + r * 32))[u] = z;
        ((uint4*)(k_sh + r * 32))[u] = z;
      }
      bf16_t zb = to_bf16(0.f);
#pragma unroll
      for (int d = 0; d < 32; ++d) vT_sh[d * 72 + r] = zb;
    }
  }
  wait_asynccnt0();
  __syncthreads();

  const float scale = 0.17677669529663687f;  // 1/sqrt(32)
  // S = q * k^T : B operand columns are k rows -> row-fragments on k_sh.
#pragma unroll
  for (int nt = 0; nt < 4; ++nt) {
    v16bf aq = load_frag_row(q_sh + nt * 16 * 32, 32);
#pragma unroll
    for (int mt = 0; mt < 4; ++mt) {
      v16bf bk = load_frag_row(k_sh + mt * 16 * 32, 32);
      v8f sacc = {};
      sacc = WMMA_BF16(aq, bk, sacc);
#pragma unroll
      for (int v = 0; v < 8; ++v) {
        int n1 = nt * 16 + v + half * 8;
        int n2 = mt * 16 + lc;
        s_sh[n1 * 65 + n2] = sacc[v] * scale + bmp[n1 * 64 + n2];
      }
    }
  }
  __syncthreads();

  // row softmax over 49 valid columns -> bf16 probabilities in p_sh
  for (int r = lane; r < 64; r += 32) {
    if (r < 49) {
      float mx = -1e30f;
      for (int c = 0; c < 49; ++c) mx = fmaxf(mx, s_sh[r * 65 + c]);
      float sum = 0.f;
      for (int c = 0; c < 49; ++c) {
        float e = __expf(s_sh[r * 65 + c] - mx);
        s_sh[r * 65 + c] = e;
        sum += e;
      }
      float inv = 1.0f / sum;
      for (int c = 0; c < 49; ++c) p_sh[r * 64 + c] = to_bf16(s_sh[r * 65 + c] * inv);
      for (int c = 49; c < 64; ++c) p_sh[r * 64 + c] = to_bf16(0.f);
    } else {
      for (int c = 0; c < 64; ++c) p_sh[r * 64 + c] = to_bf16(0.f);
    }
  }
  __syncthreads();

  // O = P * V  (K = 64 -> two k-steps of 32); V^T rows are output columns.
#pragma unroll
  for (int nt = 0; nt < 4; ++nt) {
    v16bf a0 = load_frag_row(p_sh + nt * 16 * 64, 64);
    v16bf a1 = load_frag_row(p_sh + nt * 16 * 64 + 32, 64);
#pragma unroll
    for (int dt = 0; dt < 2; ++dt) {
      v16bf b0 = load_frag_row(vT_sh + (dt * 16) * 72, 72);       // k 0..31
      v16bf b1 = load_frag_row(vT_sh + (dt * 16) * 72 + 32, 72);  // k 32..63
      v8f o = {};
      o = WMMA_BF16(a0, b0, o);
      o = WMMA_BF16(a1, b1, o);
#pragma unroll
      for (int v = 0; v < 8; ++v) {
        int n = nt * 16 + v + half * 8;
        int d = dt * 16 + lc;
        if (n < 49)
          out[(size_t)(wb * 49 + n) * 512 + head * 32 + d] = to_bf16(o[v]);
      }
    }
  }
}

// ---------- launch ----------
extern "C" void kernel_launch(void* const* d_in, const int* in_sizes, int n_in,
                              void* d_out, int out_size, void* d_ws, size_t ws_size,
                              hipStream_t stream) {
  (void)in_sizes; (void)n_in; (void)out_size; (void)ws_size;
  const float* x       = (const float*)d_in[0];
  const float* norm1_w = (const float*)d_in[1];
  const float* norm1_b = (const float*)d_in[2];
  const float* qkv_w   = (const float*)d_in[3];
  const float* qkv_b   = (const float*)d_in[4];
  const float* btab    = (const float*)d_in[5];
  const float* proj_w  = (const float*)d_in[6];
  const float* proj_b  = (const float*)d_in[7];
  const float* norm2_w = (const float*)d_in[8];
  const float* norm2_b = (const float*)d_in[9];
  const float* fc1_w   = (const float*)d_in[10];
  const float* fc1_b   = (const float*)d_in[11];
  const float* fc2_w   = (const float*)d_in[12];
  const float* fc2_b   = (const float*)d_in[13];

  const size_t T = 100352;  // 32 * 56 * 56 tokens
  size_t off = 0;
  auto alloc = [&](size_t bytes) -> void* {
    void* p = (char*)d_ws + off;
    off += (bytes + 255) & ~(size_t)255;
    return p;
  };
  bf16_t* wqkv  = (bf16_t*)alloc(512ull * 1536 * 2);
  bf16_t* wproj = (bf16_t*)alloc(512ull * 512 * 2);
  bf16_t* wfc1  = (bf16_t*)alloc(512ull * 2048 * 2);
  bf16_t* wfc2  = (bf16_t*)alloc(2048ull * 512 * 2);
  float*  bm    = (float*)alloc(4ull * 16 * 64 * 64 * 4);
  bf16_t* bufA  = (bf16_t*)alloc(T * 2048 * 2);   // qkv out / mlp hidden
  bf16_t* bufB  = (bf16_t*)alloc(T * 512 * 2);    // xw / attn-out / xn2
  float*  x1    = (float*)alloc(T * 512 * 4);     // x after attention residual

  cvt_kernel<<<(512 * 1536 + 255) / 256, 256, 0, stream>>>(qkv_w, wqkv, 512 * 1536);
  cvt_kernel<<<(512 * 512 + 255) / 256, 256, 0, stream>>>(proj_w, wproj, 512 * 512);
  cvt_kernel<<<(512 * 2048 + 255) / 256, 256, 0, stream>>>(fc1_w, wfc1, 512 * 2048);
  cvt_kernel<<<(2048 * 512 + 255) / 256, 256, 0, stream>>>(fc2_w, wfc2, 2048 * 512);
  build_bm_kernel<<<(4 * 16 * 64 * 64) / 256, 256, 0, stream>>>(btab, bm);

  // LN1 + shift + window-partition (gather), bf16 rows in window order
  ln_kernel<<<(int)T, 256, 0, stream>>>(x, norm1_w, norm1_b, bufB, 1);
  // QKV projection
  gemm_bf16_kernel<0><<<dim3(T / 128, 1536 / 128), 256, 0, stream>>>(
      bufB, wqkv, qkv_b, (int)T, 1536, 512, bufA, nullptr, nullptr);
  // windowed attention (2048 windows x 16 heads)
  attn_kernel<<<2048 * 16, 32, 0, stream>>>(bufA, bm, bufB);
  // output projection + window-reverse + unshift + residual
  gemm_bf16_kernel<2><<<dim3(T / 128, 512 / 128), 256, 0, stream>>>(
      bufB, wproj, proj_b, (int)T, 512, 512, nullptr, x1, x);
  // LN2
  ln_kernel<<<(int)T, 256, 0, stream>>>(x1, norm2_w, norm2_b, bufB, 0);
  // FC1 + GELU
  gemm_bf16_kernel<1><<<dim3(T / 128, 2048 / 128), 256, 0, stream>>>(
      bufB, wfc1, fc1_b, (int)T, 2048, 512, bufA, nullptr, nullptr);
  // FC2 + residual -> output
  gemm_bf16_kernel<3><<<dim3(T / 128, 512 / 128), 256, 0, stream>>>(
      bufA, wfc2, fc2_b, (int)T, 512, 2048, nullptr, (float*)d_out, x1);
}
